// MambaLayer_36386962932125
// MI455X (gfx1250) — compile-verified, matrix operand order checked by reading, not run on hardware
//
#include <hip/hip_runtime.h>

// ---------------------------------------------------------------------------
// Mamba2 layer forward for MI455X (gfx1250, wave32, WMMA + TDM).
// Pipeline: cast->bf16 (zero-padded N), WMMA GEMM with TDM-staged LDS slabs
// (in_proj), conv1d+SiLU, sequential SSM scan, gate+RMSNorm (emit bf16),
// WMMA GEMM (out_proj).
// ---------------------------------------------------------------------------

typedef __attribute__((ext_vector_type(16))) __bf16 v16bf;
typedef __attribute__((ext_vector_type(8)))  float  v8f;
typedef __attribute__((ext_vector_type(4)))  unsigned int u32x4;
typedef __attribute__((ext_vector_type(8)))  int i32x8;
typedef __attribute__((ext_vector_type(4)))  int i32x4;

union BF16Tile { v16bf v; u32x4 u[2]; };

#if __has_builtin(__builtin_amdgcn_tensor_load_to_lds) && \
    __has_builtin(__builtin_amdgcn_s_wait_tensorcnt)
#define USE_TDM 1
#else
#define USE_TDM 0
#endif

constexpr int BATCH   = 2;
constexpr int LSEQ    = 4096;
constexpr int DMODEL  = 1024;
constexpr int DINNER  = 2048;
constexpr int HEADDIM = 64;
constexpr int NHEADS  = 32;
constexpr int DSTATE  = 128;
constexpr int DCONV   = 4;
constexpr int DOUT    = 1024;
constexpr int DPROJ   = 2 * DINNER + 2 * DSTATE + NHEADS;  // 4384
constexpr int DPROJP  = 4480;                              // padded to 35*128
constexpr int CONVDIM = DINNER + 2 * DSTATE;               // 2304
constexpr int MROWS   = BATCH * LSEQ;                      // 8192
constexpr float EPSF  = 1e-5f;

// GEMM tiling
constexpr int BM = 128, BN = 128, BK = 64;    // block tile
constexpr int LDSROW = 72;   // padded LDS row (elems): 128B data + 16B pad

__device__ __forceinline__ unsigned short f2bf(float f) {
    unsigned int u = __float_as_uint(f);
    unsigned int r = u + 0x7FFFu + ((u >> 16) & 1u);   // round-to-nearest-even
    return (unsigned short)(r >> 16);
}

__device__ __forceinline__ unsigned lds_off(const void* p) {
    // LDS aperture maps addr[31:0] directly to the LDS byte offset (ISA 10.2)
    return (unsigned)(unsigned long long)p;
}

#if USE_TDM
// Issue a TDM load of a [BM rows x BK cols] bf16 tile (row stride K elems)
// into LDS with 16B padding after each 128B row (-> LDSROW elem stride).
__device__ __forceinline__ void tdm_load_tile(const unsigned short* gsrc,
                                              unsigned ldsOff, int K) {
    const unsigned long long ga = (unsigned long long)gsrc;
    u32x4 g0;
    g0[0] = 1u;                                       // count=1, user mode
    g0[1] = ldsOff;                                   // lds_addr
    g0[2] = (unsigned)(ga & 0xffffffffu);             // global_addr[31:0]
    g0[3] = (unsigned)((ga >> 32) & 0x01ffffffu)      // global_addr[56:32]
            | (2u << 30);                             // type = 2 ("image")
    i32x8 g1;
    g1[0] = (1 << 16)      // data_size = 2 bytes
          | (1 << 20)      // pad_enable
          | (4 << 22)      // pad_interval: 32 DWORDs (=128B) before pad
          | (3 << 25);     // pad_amount:   4 DWORDs (=16B)
    g1[1] = (BK & 0xffff) << 16;                      // tensor_dim0 = BK
    g1[2] = (BM & 0xffff) << 16;                      // tensor_dim1 = BM
    g1[3] = (BK & 0xffff) << 16;                      // tile_dim0 = BK
    g1[4] = BM;                                       // tile_dim1 = BM
    g1[5] = K;                                        // tensor_dim0_stride
    g1[6] = 0;
    g1[7] = 0;
    const i32x4 gz4 = {0, 0, 0, 0};                   // 2D: groups 2/3 unused
    const i32x8 gz8 = {0, 0, 0, 0, 0, 0, 0, 0};
    __builtin_amdgcn_tensor_load_to_lds(g0, g1, gz4, gz4, gz8, 0);
}
#endif

// --------------- fp32 -> bf16 cast with zero-fill tail padding -------------
__global__ void cast_f32_bf16_pad(const float* __restrict__ in,
                                  unsigned short* __restrict__ out,
                                  long nvalid, long ntotal) {
    long i = (long)blockIdx.x * blockDim.x + threadIdx.x;
    if (i < ntotal) out[i] = (i < nvalid) ? f2bf(in[i]) : (unsigned short)0;
}

// ------------------- bf16 WMMA GEMM:  C = A * B^T (f32 acc) ----------------
// A: [M x K] bf16, Bw: [N x K] bf16 (N mult of 128, zero-padded rows),
// C: [M x N] f32.  256 threads = 8 waves; block tile 128x128, K staged in
// 64-wide double-buffered LDS slabs moved by the Tensor Data Mover.
// Each wave computes 32(M) x 64(N) => 8 f32 accumulators.
__global__ void __launch_bounds__(256)
gemm_bf16_nt(const unsigned short* __restrict__ A,
             const unsigned short* __restrict__ Bw,
             float* __restrict__ C,
             int M, int N, int K) {
    __shared__ __align__(16) unsigned short sA[2][BM * LDSROW];
    __shared__ __align__(16) unsigned short sB[2][BN * LDSROW];

    const int tid  = threadIdx.x;
    const int wave = tid >> 5;
    const int lane = tid & 31;
    const int r    = lane & 15;      // row (A frag) / col (B frag) in subtile
    const int kh   = lane >> 4;      // K half-select (ISA 7.12.2 layout)
    const int wm   = (wave >> 1) * 2;  // first of 2 M-subtiles for this wave
    const int wn   = (wave & 1) * 4;   // first of 4 N-subtiles for this wave

    const int m0 = blockIdx.y * BM;
    const int n0 = blockIdx.x * BN;
    const int nst = K / BK;

#if !USE_TDM
    int srow[4], skc[4];
#pragma unroll
    for (int q = 0; q < 4; ++q) {
        const int c = tid + q * 256;
        srow[q] = c >> 3;
        skc[q]  = (c & 7) * 8;
    }
    u32x4 ra[4], rb[4];
#endif

    // ---- preload stage 0 into buffer 0 ----
#if USE_TDM
    if (wave == 0) {
        tdm_load_tile(A  + (size_t)m0 * K, lds_off(&sA[0][0]), K);
        tdm_load_tile(Bw + (size_t)n0 * K, lds_off(&sB[0][0]), K);
        __builtin_amdgcn_s_wait_tensorcnt((short)0);
    }
#else
#pragma unroll
    for (int q = 0; q < 4; ++q) {
        ra[q] = *(const u32x4*)(A  + (size_t)(m0 + srow[q]) * K + skc[q]);
        rb[q] = *(const u32x4*)(Bw + (size_t)(n0 + srow[q]) * K + skc[q]);
    }
#pragma unroll
    for (int q = 0; q < 4; ++q) {
        *(u32x4*)(&sA[0][srow[q] * LDSROW + skc[q]]) = ra[q];
        *(u32x4*)(&sB[0][srow[q] * LDSROW + skc[q]]) = rb[q];
    }
#endif
    __syncthreads();   // stage-0 slab visible to all waves

    v8f acc[2][4] = {};

    for (int s = 0; s < nst; ++s) {
        const int buf = s & 1;

        // kick off next-stage staging (other buffer: no race with compute)
        if (s + 1 < nst) {
            const int k0 = (s + 1) * BK;
#if USE_TDM
            if (wave == 0) {
                tdm_load_tile(A  + (size_t)m0 * K + k0, lds_off(&sA[buf ^ 1][0]), K);
                tdm_load_tile(Bw + (size_t)n0 * K + k0, lds_off(&sB[buf ^ 1][0]), K);
            }
#else
#pragma unroll
            for (int q = 0; q < 4; ++q) {
                ra[q] = *(const u32x4*)(A  + (size_t)(m0 + srow[q]) * K + k0 + skc[q]);
                rb[q] = *(const u32x4*)(Bw + (size_t)(n0 + srow[q]) * K + k0 + skc[q]);
            }
#endif
        }

        // compute: two 16x16x32 WMMA k-steps from the LDS slab
#pragma unroll
        for (int kk2 = 0; kk2 < 2; ++kk2) {
            const int kk = kk2 * 32;
            BF16Tile af[2], bfr[4];
#pragma unroll
            for (int ms = 0; ms < 2; ++ms) {
                const unsigned short* base =
                    &sA[buf][((wm + ms) * 16 + r) * LDSROW + kk + kh * 8];
                af[ms].u[0] = *(const u32x4*)(base);
                af[ms].u[1] = *(const u32x4*)(base + 16);
            }
#pragma unroll
            for (int ns = 0; ns < 4; ++ns) {
                const unsigned short* base =
                    &sB[buf][((wn + ns) * 16 + r) * LDSROW + kk + kh * 8];
                bfr[ns].u[0] = *(const u32x4*)(base);
                bfr[ns].u[1] = *(const u32x4*)(base + 16);
            }
#pragma unroll
            for (int ms = 0; ms < 2; ++ms)
#pragma unroll
                for (int ns = 0; ns < 4; ++ns)
                    acc[ms][ns] = __builtin_amdgcn_wmma_f32_16x16x32_bf16(
                        false, af[ms].v, false, bfr[ns].v, (short)0,
                        acc[ms][ns], false, false);
        }

        // publish next slab and re-converge
        if (s + 1 < nst) {
#if USE_TDM
            if (wave == 0) __builtin_amdgcn_s_wait_tensorcnt((short)0);
#else
            const int nb = buf ^ 1;
#pragma unroll
            for (int q = 0; q < 4; ++q) {
                *(u32x4*)(&sA[nb][srow[q] * LDSROW + skc[q]]) = ra[q];
                *(u32x4*)(&sB[nb][srow[q] * LDSROW + skc[q]]) = rb[q];
            }
#endif
            __syncthreads();
        }
    }

    // store: D layout (ISA 7.12.2): elem e -> row e + 8*kh, col r
#pragma unroll
    for (int ms = 0; ms < 2; ++ms) {
        const int mrow = m0 + (wm + ms) * 16 + 8 * kh;
#pragma unroll
        for (int ns = 0; ns < 4; ++ns) {
            const int n = n0 + (wn + ns) * 16 + r;
#pragma unroll
            for (int e = 0; e < 8; ++e)
                C[(size_t)(mrow + e) * N + n] = acc[ms][ns][e];
        }
    }
}

// ------------------ depthwise causal conv1d (k=4) + SiLU -------------------
__global__ void conv_silu(const float* __restrict__ zx,
                          const float* __restrict__ cw,
                          const float* __restrict__ cb,
                          float* __restrict__ outbuf, long total) {
    long idx = (long)blockIdx.x * blockDim.x + threadIdx.x;
    if (idx >= total) return;
    const int  c  = (int)(idx % CONVDIM);
    const long bl = idx / CONVDIM;
    const int  l  = (int)(bl % LSEQ);
    const long b0 = bl - l;  // b * LSEQ
    const float* col = zx + (size_t)b0 * DPROJP + DINNER + c;
    float acc = cb[c];
#pragma unroll
    for (int j = 0; j < DCONV; ++j) {
        const int lj = l - (DCONV - 1) + j;
        if (lj >= 0) acc += cw[c * DCONV + j] * col[(size_t)lj * DPROJP];
    }
    outbuf[idx] = acc / (1.f + __expf(-acc));   // SiLU
}

// ----------------------------- SSM scan ------------------------------------
// One block per (batch, head). State h[64][128]: thread (i = tid>>2) owns 32
// state columns; C-dot reduced across the 4 co-lanes with __shfl_xor (wave32).
__global__ void __launch_bounds__(256)
ssm_scan(const float* __restrict__ convout,
         const float* __restrict__ zx,
         const float* __restrict__ dt_bias,
         const float* __restrict__ A_log,
         const float* __restrict__ Dp,
         float* __restrict__ yscan) {
    const int b   = blockIdx.x / NHEADS;
    const int h   = blockIdx.x % NHEADS;
    const int tid = threadIdx.x;
    const int i   = tid >> 2;      // headdim index 0..63
    const int g   = tid & 3;       // state-group 0..3
    const int j0  = g * 32;

    __shared__ float sB[DSTATE], sC[DSTATE], sx[HEADDIM], sScal[2];

    const float Ah   = -__expf(A_log[h]);
    const float bias = dt_bias[h];
    const float Dh   = Dp[h];

    float hst[32];
#pragma unroll
    for (int jj = 0; jj < 32; ++jj) hst[jj] = 0.f;

    for (int t = 0; t < LSEQ; ++t) {
        const float* row = convout + ((size_t)b * LSEQ + t) * CONVDIM;
        if (tid < DSTATE) sB[tid] = row[DINNER + tid];
        else              sC[tid - DSTATE] = row[DINNER + tid];
        if (tid < HEADDIM) sx[tid] = row[h * HEADDIM + tid];
        if (tid == 0) {
            float draw = zx[((size_t)b * LSEQ + t) * DPROJP + (DINNER + CONVDIM) + h] + bias;
            float dtv  = (draw > 20.f) ? draw : log1pf(__expf(draw));  // softplus
            sScal[0] = dtv;
            sScal[1] = __expf(dtv * Ah);
        }
        __syncthreads();

        const float dtv = sScal[0], dAv = sScal[1];
        const float xi   = sx[i];
        const float coef = dtv * xi;
        float acc = 0.f;
#pragma unroll
        for (int jj = 0; jj < 32; ++jj) {
            const int j = j0 + jj;
            hst[jj] = hst[jj] * dAv + coef * sB[j];
            acc    += hst[jj] * sC[j];
        }
        acc += __shfl_xor(acc, 1, 32);
        acc += __shfl_xor(acc, 2, 32);
        if (g == 0)
            yscan[((size_t)b * LSEQ + t) * DINNER + h * HEADDIM + i] = acc + Dh * xi;
        __syncthreads();
    }
}

// --------------------- gate (SiLU(z)) + RMSNorm -> bf16 --------------------
__global__ void __launch_bounds__(256)
gate_rmsnorm(const float* __restrict__ yscan,
             const float* __restrict__ zx,
             const float* __restrict__ norm_w,
             unsigned short* __restrict__ ybf) {
    const int row = blockIdx.x;
    const int tid = threadIdx.x;
    __shared__ float vals[DINNER];
    __shared__ float red[256];

    const float* y = yscan + (size_t)row * DINNER;
    const float* z = zx    + (size_t)row * DPROJP;  // z = cols [0, DINNER)

    float ss = 0.f;
#pragma unroll
    for (int q = 0; q < DINNER / 256; ++q) {
        const int c = q * 256 + tid;
        const float zv = z[c];
        const float v  = y[c] * (zv / (1.f + __expf(-zv)));
        vals[c] = v;
        ss += v * v;
    }
    red[tid] = ss;
    __syncthreads();
    for (int s = 128; s > 0; s >>= 1) {
        if (tid < s) red[tid] += red[tid + s];
        __syncthreads();
    }
    const float scale = rsqrtf(red[0] / DINNER + EPSF);
#pragma unroll
    for (int q = 0; q < DINNER / 256; ++q) {
        const int c = q * 256 + tid;
        ybf[(size_t)row * DINNER + c] = f2bf(vals[c] * scale * norm_w[c]);
    }
}

// ---------------------------------------------------------------------------
extern "C" void kernel_launch(void* const* d_in, const int* in_sizes, int n_in,
                              void* d_out, int out_size, void* d_ws, size_t ws_size,
                              hipStream_t stream) {
    (void)in_sizes; (void)n_in; (void)out_size; (void)ws_size;
    const float* x       = (const float*)d_in[0];
    const float* w_in    = (const float*)d_in[1];
    const float* conv_w  = (const float*)d_in[2];
    const float* conv_b  = (const float*)d_in[3];
    const float* dt_bias = (const float*)d_in[4];
    const float* A_log   = (const float*)d_in[5];
    const float* Dp      = (const float*)d_in[6];
    const float* norm_w  = (const float*)d_in[7];
    const float* w_out   = (const float*)d_in[8];
    float* out = (float*)d_out;

    char* ws = (char*)d_ws;
    size_t off = 0;
    auto carve = [&](size_t bytes) {
        void* p = ws + off;
        off += (bytes + 255) & ~(size_t)255;
        return p;
    };
    unsigned short* xbf  = (unsigned short*)carve((size_t)MROWS  * DMODEL * 2);
    unsigned short* wibf = (unsigned short*)carve((size_t)DPROJP * DMODEL * 2);  // zero-padded rows
    unsigned short* wobf = (unsigned short*)carve((size_t)DOUT   * DINNER * 2);
    float*          zx   = (float*)carve((size_t)MROWS * DPROJP  * 4);
    float*          conv = (float*)carve((size_t)MROWS * CONVDIM * 4);
    float*          ysc  = (float*)carve((size_t)MROWS * DINNER  * 4);
    unsigned short* ybf  = (unsigned short*)carve((size_t)MROWS * DINNER * 2);

    {   // casts to bf16 (weights padded with zero rows up to DPROJP)
        long n = (long)MROWS * DMODEL;
        cast_f32_bf16_pad<<<(n + 255) / 256, 256, 0, stream>>>(x, xbf, n, n);
        long nv = (long)DPROJ * DMODEL, nt = (long)DPROJP * DMODEL;
        cast_f32_bf16_pad<<<(nt + 255) / 256, 256, 0, stream>>>(w_in, wibf, nv, nt);
        n = (long)DOUT * DINNER;
        cast_f32_bf16_pad<<<(n + 255) / 256, 256, 0, stream>>>(w_out, wobf, n, n);
    }

    // in_proj: zx[8192 x 4480p] = x_bf16[8192 x 1024] * w_in^T   (guard-free)
    {
        dim3 grid(DPROJP / BN, MROWS / BM);
        gemm_bf16_nt<<<grid, 256, 0, stream>>>(xbf, wibf, zx, MROWS, DPROJP, DMODEL);
    }

    // causal depthwise conv + SiLU
    {
        long n = (long)MROWS * CONVDIM;
        conv_silu<<<(n + 255) / 256, 256, 0, stream>>>(zx, conv_w, conv_b, conv, n);
    }

    // sequential SSM scan: one block per (batch, head)
    ssm_scan<<<BATCH * NHEADS, 256, 0, stream>>>(conv, zx, dt_bias, A_log, Dp, ysc);

    // gating + RMSNorm, emit bf16 activations
    gate_rmsnorm<<<MROWS, 256, 0, stream>>>(ysc, zx, norm_w, ybf);

    // out_proj: out[8192 x 1024] = y_bf16[8192 x 2048] * w_out^T
    {
        dim3 grid(DOUT / BN, MROWS / BM);
        gemm_bf16_nt<<<grid, 256, 0, stream>>>(ybf, wobf, out, MROWS, DOUT, DINNER);
    }
}
